// LinearCausalAttention_46179488366615
// MI455X (gfx1250) — matrix operand order repaired and checked by reading, not compiled, not run on hardware
//
#include <hip/hip_runtime.h>

typedef _Float16 f16;
typedef __attribute__((ext_vector_type(16))) _Float16 v16h;
typedef __attribute__((ext_vector_type(8)))  _Float16 v8h;
typedef __attribute__((ext_vector_type(4)))  _Float16 v4h;
typedef __attribute__((ext_vector_type(8)))  float    v8f;

#define S_LEN 2048
#define DMODEL 1024
#define NHEAD 16
#define HD 64
#define EPSV 1e-6f

static __device__ __forceinline__ v8f zero8() {
  v8f z;
#pragma unroll
  for (int i = 0; i < 8; i++) z[i] = 0.f;
  return z;
}

static __device__ __forceinline__ v8f wmma(v16h a, v16h b, v8f c) {
  return __builtin_amdgcn_wmma_f32_16x16x32_f16(false, a, false, b, (short)0, c, false, false);
}

static __device__ __forceinline__ v16h ld2x8(const f16* p0, const f16* p1) {
  v8h a = *(const v8h*)p0;
  v8h b = *(const v8h*)p1;
  v16h f;
#pragma unroll
  for (int e = 0; e < 8; e++) { f[e] = a[e]; f[8 + e] = b[e]; }
  return f;
}

// A fragment from row-major [M][K] tile (ld % 8 == 0): 2x ds_load_b128
static __device__ __forceinline__ v16h frag_a(const f16* p, int ld, int row0, int k0) {
  int lane = threadIdx.x & 31, l = lane & 15, hi = lane >> 4;
  const f16* base = p + (row0 + l) * ld + k0 + hi * 8;
  return ld2x8(base, base + 16);
}

// B fragment from column-major [N][K] tile (ld % 8 == 0): 2x ds_load_b128
static __device__ __forceinline__ v16h frag_b_cm(const f16* p, int ld, int k0, int n0) {
  int lane = threadIdx.x & 31, n = lane & 15, hi = lane >> 4;
  const f16* base = p + (n0 + n) * ld + k0 + hi * 8;
  return ld2x8(base, base + 16);
}

static __device__ __forceinline__ float phif(float x) {
  return x > 0.f ? x + 1.f : __expf(x);  // elu(x)+1
}

// ---------------------------------------------------------------------------
// Kernel 1: qkv = x @ W_qkv (2048x1024x3072). 256 thr / 8 waves, 128x128 tile,
// BK=32, double-buffered LDS, float4 global loads, fused feature map.
// ---------------------------------------------------------------------------
__global__ __launch_bounds__(256) void qkv_gemm(const float* __restrict__ X,
                                                const float* __restrict__ W,
                                                f16* __restrict__ phiQ,
                                                f16* __restrict__ phiK,
                                                f16* __restrict__ Vb) {
  const int LD = 48;
  const int NN = 3072;
  __shared__ __align__(16) f16 sA[2][128 * 48];
  __shared__ __align__(16) f16 sB[2][128 * 48];
  int tid = threadIdx.x;
  int w = tid >> 5, lane = tid & 31, n = lane & 15, hi = lane >> 4;
  int wi = w & 3, wj = w >> 2;
  int m0 = blockIdx.y * 128, n0 = blockIdx.x * 128;

  // per-thread staging bases (all later offsets are immediates)
  const float* gA = X + (size_t)(m0 + (tid >> 3)) * DMODEL + (tid & 7) * 4;
  const float* gB = W + (size_t)(tid >> 5) * NN + n0 + (tid & 31) * 4;

  v8f acc[2][4];
#pragma unroll
  for (int s = 0; s < 2; s++)
#pragma unroll
    for (int j = 0; j < 4; j++) acc[s][j] = zero8();

  float4 ra[4], rb[4];
  auto load_stage = [&](int k0) {
#pragma unroll
    for (int e = 0; e < 4; e++) {
      ra[e] = *(const float4*)(gA + (size_t)e * 32 * DMODEL + k0);
      rb[e] = *(const float4*)(gB + (size_t)((size_t)e * 8 + k0) * NN);
    }
  };
  auto store_stage = [&](int b) {
#pragma unroll
    for (int e = 0; e < 4; e++) {
      v4h ha;
      ha[0] = (f16)ra[e].x; ha[1] = (f16)ra[e].y;
      ha[2] = (f16)ra[e].z; ha[3] = (f16)ra[e].w;
      *(v4h*)&sA[b][(e * 32 + (tid >> 3)) * LD + (tid & 7) * 4] = ha;
      int rr = e * 8 + (tid >> 5);
      int cb = (tid & 31) * 4;
      sB[b][(cb + 0) * LD + rr] = (f16)rb[e].x;
      sB[b][(cb + 1) * LD + rr] = (f16)rb[e].y;
      sB[b][(cb + 2) * LD + rr] = (f16)rb[e].z;
      sB[b][(cb + 3) * LD + rr] = (f16)rb[e].w;
    }
  };

  load_stage(0);
  store_stage(0);
  __syncthreads();

  for (int i = 0; i < 32; i++) {
    int buf = i & 1;
    if (i + 1 < 32) load_stage((i + 1) * 32);  // issue loads, no wait yet
    v16h a0 = frag_a(sA[buf], LD, 32 * wi, 0);
    v16h a1 = frag_a(sA[buf], LD, 32 * wi + 16, 0);
#pragma unroll
    for (int j = 0; j < 4; j++) {
      v16h b = frag_b_cm(sB[buf], LD, 0, 64 * wj + 16 * j);
      acc[0][j] = wmma(a0, b, acc[0][j]);
      acc[1][j] = wmma(a1, b, acc[1][j]);
    }
    if (i + 1 < 32) {
      store_stage(1 - buf);  // waits loadcnt here, overlapped with wmmas
      __syncthreads();
    }
  }

  // epilogue: part & head are wave-uniform
  int part = n0 >> 10;                          // 0=q 1=k 2=v
  int hh = ((n0 & 1023) + 64 * wj) >> 6;        // head id
  f16* dst = (part == 0) ? phiQ : (part == 1) ? phiK : Vb;
  dst += (size_t)hh * S_LEN * HD;
  if (part < 2) {
#pragma unroll
    for (int s = 0; s < 2; s++)
#pragma unroll
      for (int j = 0; j < 4; j++)
#pragma unroll
        for (int r = 0; r < 8; r++) {
          int row = m0 + 32 * wi + 16 * s + r + 8 * hi;
          dst[(size_t)row * HD + 16 * j + n] = (f16)phif(acc[s][j][r]);
        }
  } else {
#pragma unroll
    for (int s = 0; s < 2; s++)
#pragma unroll
      for (int j = 0; j < 4; j++)
#pragma unroll
        for (int r = 0; r < 8; r++) {
          int row = m0 + 32 * wi + 16 * s + r + 8 * hi;
          dst[(size_t)row * HD + 16 * j + n] = (f16)acc[s][j][r];
        }
  }
}

// ---------------------------------------------------------------------------
// Kernel 2: chunked causal linear attention. 1 workgroup per head, 8 waves.
// Chunk = 64 steps. Persistent KV state (64x64 fp32) lives in WMMA accums.
// ---------------------------------------------------------------------------
__global__ __launch_bounds__(256) void linattn(const f16* __restrict__ phiQ,
                                               const f16* __restrict__ phiK,
                                               const f16* __restrict__ Vb,
                                               f16* __restrict__ attn) {
  const int LDT = 72;
  __shared__ __align__(16) f16 sQ[64 * 64];    // phiQ chunk, row-major [s][c]
  __shared__ __align__(16) f16 sK[64 * 64];    // phiK chunk, row-major [t][c]
  __shared__ __align__(16) f16 sKt[64 * LDT];  // phiK chunk, col-major [c][t]
  __shared__ __align__(16) f16 sVt[64 * LDT];  // V chunk,    col-major [j][t]
  __shared__ __align__(16) f16 sKVt[64 * LDT]; // KV state,   col-major [j][c]
  __shared__ __align__(16) f16 sA[64 * 64];    // masked scores, row-major
  __shared__ float sDen[64], sKst[64], sKold[64];

  int tid = threadIdx.x, w = tid >> 5, lane = tid & 31, l = lane & 15, hi = lane >> 4;
  int h = blockIdx.x;
  const f16* Q = phiQ + (size_t)h * S_LEN * HD;
  const f16* K = phiK + (size_t)h * S_LEN * HD;
  const f16* V = Vb + (size_t)h * S_LEN * HD;

  int i = w >> 1;
  int j0 = (w & 1) * 2;

  v8f kv0 = zero8(), kv1 = zero8();
  if (tid < 64) sKst[tid] = 0.f;

  for (int ch = 0; ch < S_LEN / 64; ch++) {
    int s0 = ch * 64;
    __syncthreads();  // (a)

    {
      const uint4* gq = (const uint4*)(Q + (size_t)s0 * HD);
      const uint4* gk = (const uint4*)(K + (size_t)s0 * HD);
      const uint4* gv = (const uint4*)(V + (size_t)s0 * HD);
      uint4* q4 = (uint4*)sQ;
      uint4* k4 = (uint4*)sK;
      __builtin_prefetch(gq + 512, 0, 0);
      __builtin_prefetch(gk + 512, 0, 0);
      __builtin_prefetch(gv + 512, 0, 0);
#pragma unroll
      for (int e = 0; e < 2; e++) {
        int v = e * 256 + tid;
        int t = v >> 3;
        int c0 = (v & 7) * 8;
        uint4 rq = gq[v];
        uint4 rk = gk[v];
        uint4 rv = gv[v];
        q4[v] = rq;
        k4[v] = rk;
        union { uint4 u; f16 x[8]; } uk, uv;
        uk.u = rk; uv.u = rv;
#pragma unroll
        for (int u = 0; u < 8; u++) {
          sKt[(c0 + u) * LDT + t] = uk.x[u];
          sVt[(c0 + u) * LDT + t] = uv.x[u];
        }
      }
    }
#pragma unroll
    for (int r = 0; r < 8; r++) {
      int M = r + 8 * hi;
      sKVt[(16 * j0 + l) * LDT + 16 * i + M]       = (f16)kv0[r];
      sKVt[(16 * (j0 + 1) + l) * LDT + 16 * i + M] = (f16)kv1[r];
    }
    if (tid < 64) sKold[tid] = sKst[tid];
    __syncthreads();  // (b)

    v16h aq0 = frag_a(sQ, 64, 16 * i, 0);
    v16h aq1 = frag_a(sQ, 64, 16 * i, 32);

    v8f oa0 = wmma(aq0, frag_b_cm(sKVt, LDT, 0, 16 * j0), zero8());
    oa0 = wmma(aq1, frag_b_cm(sKVt, LDT, 32, 16 * j0), oa0);
    v8f oa1 = wmma(aq0, frag_b_cm(sKVt, LDT, 0, 16 * (j0 + 1)), zero8());
    oa1 = wmma(aq1, frag_b_cm(sKVt, LDT, 32, 16 * (j0 + 1)), oa1);

    v8f sc0 = wmma(aq0, frag_b_cm(sK, 64, 0, 16 * j0), zero8());
    sc0 = wmma(aq1, frag_b_cm(sK, 64, 32, 16 * j0), sc0);
    v8f sc1 = wmma(aq0, frag_b_cm(sK, 64, 0, 16 * (j0 + 1)), zero8());
    sc1 = wmma(aq1, frag_b_cm(sK, 64, 32, 16 * (j0 + 1)), sc1);

#pragma unroll
    for (int r = 0; r < 8; r++) {
      int srow = 16 * i + r + 8 * hi;
      int t0c = 16 * j0 + l;
      int t1c = 16 * (j0 + 1) + l;
      sA[srow * 64 + t0c] = (f16)((t0c <= srow) ? sc0[r] : 0.f);
      sA[srow * 64 + t1c] = (f16)((t1c <= srow) ? sc1[r] : 0.f);
    }
    __syncthreads();  // (c)

    if (tid < 64) {
      float d = EPSV;
      for (int t = 0; t < 64; t++) d += (float)sA[tid * 64 + t];
      for (int c2 = 0; c2 < 64; c2++) d += (float)sQ[tid * 64 + c2] * sKold[c2];
      sDen[tid] = d;
    }

    v16h bv00 = frag_b_cm(sVt, LDT, 0, 16 * j0);
    v16h bv10 = frag_b_cm(sVt, LDT, 32, 16 * j0);
    v16h bv01 = frag_b_cm(sVt, LDT, 0, 16 * (j0 + 1));
    v16h bv11 = frag_b_cm(sVt, LDT, 32, 16 * (j0 + 1));

    v16h am0 = frag_a(sA, 64, 16 * i, 0);
    v16h am1 = frag_a(sA, 64, 16 * i, 32);
    oa0 = wmma(am0, bv00, oa0); oa0 = wmma(am1, bv10, oa0);
    oa1 = wmma(am0, bv01, oa1); oa1 = wmma(am1, bv11, oa1);

    v16h at0 = frag_a(sKt, LDT, 16 * i, 0);
    v16h at1 = frag_a(sKt, LDT, 16 * i, 32);
    kv0 = wmma(at0, bv00, kv0); kv0 = wmma(at1, bv10, kv0);
    kv1 = wmma(at0, bv01, kv1); kv1 = wmma(at1, bv11, kv1);

    __syncthreads();  // (d)

#pragma unroll
    for (int r = 0; r < 8; r++) {
      int srow = 16 * i + r + 8 * hi;
      float den = sDen[srow];
      size_t base = (size_t)(s0 + srow) * DMODEL + h * HD;
      attn[base + 16 * j0 + l]       = (f16)(oa0[r] / den);
      attn[base + 16 * (j0 + 1) + l] = (f16)(oa1[r] / den);
    }
    if (tid < 64) {
      float a = sKst[tid];
      for (int t = 0; t < 64; t++) a += (float)sKt[tid * LDT + t];
      sKst[tid] = a;
    }
  }
}

// ---------------------------------------------------------------------------
// Kernel 3: out = attn(f16) @ W_out (2048x1024x1024), fp32 result.
// Same 128x128 double-buffered structure as qkv_gemm.
// ---------------------------------------------------------------------------
__global__ __launch_bounds__(256) void out_gemm(const f16* __restrict__ A,
                                                const float* __restrict__ W,
                                                float* __restrict__ Out) {
  const int LD = 48;
  const int NN = 1024;
  __shared__ __align__(16) f16 sA[2][128 * 48];
  __shared__ __align__(16) f16 sB[2][128 * 48];
  int tid = threadIdx.x;
  int w = tid >> 5, lane = tid & 31, n = lane & 15, hi = lane >> 4;
  int wi = w & 3, wj = w >> 2;
  int m0 = blockIdx.y * 128, n0 = blockIdx.x * 128;

  const f16* gA = A + (size_t)(m0 + (tid >> 2)) * DMODEL + (tid & 3) * 8;
  const float* gB = W + (size_t)(tid >> 5) * NN + n0 + (tid & 31) * 4;

  v8f acc[2][4];
#pragma unroll
  for (int s = 0; s < 2; s++)
#pragma unroll
    for (int j = 0; j < 4; j++) acc[s][j] = zero8();

  uint4 ra[2];
  float4 rb[4];
  auto load_stage = [&](int k0) {
#pragma unroll
    for (int e = 0; e < 2; e++)
      ra[e] = *(const uint4*)(gA + (size_t)e * 64 * DMODEL + k0);
#pragma unroll
    for (int e = 0; e < 4; e++)
      rb[e] = *(const float4*)(gB + (size_t)((size_t)e * 8 + k0) * NN);
  };
  auto store_stage = [&](int b) {
#pragma unroll
    for (int e = 0; e < 2; e++)
      *(uint4*)&sA[b][(e * 64 + (tid >> 2)) * LD + (tid & 3) * 8] = ra[e];
#pragma unroll
    for (int e = 0; e < 4; e++) {
      int rr = e * 8 + (tid >> 5);
      int cb = (tid & 31) * 4;
      sB[b][(cb + 0) * LD + rr] = (f16)rb[e].x;
      sB[b][(cb + 1) * LD + rr] = (f16)rb[e].y;
      sB[b][(cb + 2) * LD + rr] = (f16)rb[e].z;
      sB[b][(cb + 3) * LD + rr] = (f16)rb[e].w;
    }
  };

  load_stage(0);
  store_stage(0);
  __syncthreads();

  for (int i = 0; i < 32; i++) {
    int buf = i & 1;
    if (i + 1 < 32) load_stage((i + 1) * 32);
    v16h a0 = frag_a(sA[buf], LD, 32 * wi, 0);
    v16h a1 = frag_a(sA[buf], LD, 32 * wi + 16, 0);
#pragma unroll
    for (int j = 0; j < 4; j++) {
      v16h b = frag_b_cm(sB[buf], LD, 0, 64 * wj + 16 * j);
      acc[0][j] = wmma(a0, b, acc[0][j]);
      acc[1][j] = wmma(a1, b, acc[1][j]);
    }
    if (i + 1 < 32) {
      store_stage(1 - buf);
      __syncthreads();
    }
  }

#pragma unroll
  for (int s = 0; s < 2; s++)
#pragma unroll
    for (int j = 0; j < 4; j++)
#pragma unroll
      for (int r = 0; r < 8; r++) {
        int row = m0 + 32 * wi + 16 * s + r + 8 * hi;
        int col = n0 + 64 * wj + 16 * j + n;
        Out[(size_t)row * DMODEL + col] = acc[s][j][r];
      }
}

extern "C" void kernel_launch(void* const* d_in, const int* in_sizes, int n_in,
                              void* d_out, int out_size, void* d_ws, size_t ws_size,
                              hipStream_t stream) {
  (void)in_sizes; (void)n_in; (void)out_size; (void)ws_size;
  const float* x    = (const float*)d_in[0];
  // d_in[1] = causal_mask (implicit in the chunked recurrence; unused)
  const float* Wqkv = (const float*)d_in[2];
  const float* Wout = (const float*)d_in[3];
  float* out = (float*)d_out;

  const size_t HSD = (size_t)NHEAD * S_LEN * HD;  // 2M elements
  f16* phiQ = (f16*)d_ws;
  f16* phiK = phiQ + HSD;
  f16* Vb   = phiK + HSD;
  f16* attn = Vb + HSD;   // (S, D) f16

  qkv_gemm<<<dim3(3072 / 128, S_LEN / 128), 256, 0, stream>>>(x, Wqkv, phiQ, phiK, Vb);
  linattn<<<dim3(NHEAD), 256, 0, stream>>>(phiQ, phiK, Vb, attn);
  out_gemm<<<dim3(DMODEL / 128, S_LEN / 128), 256, 0, stream>>>(attn, Wout, out);
}
